// GAT_63660005261515
// MI455X (gfx1250) — compile-verified, matrix operand order checked by reading, not run on hardware
//
#include <hip/hip_runtime.h>
#include <math.h>

// Problem constants (match reference)
#define N_NODES 100000
#define N_EDGES 800000
#define HD      128     // H*D
#define DIN     32      // node/edge input feature dim (= K of all GEMMs)
#define NLAYERS 3

typedef __attribute__((ext_vector_type(16))) _Float16 v16h;
typedef __attribute__((ext_vector_type(8)))  _Float16 v8h;
typedef __attribute__((ext_vector_type(8)))  float    v8f;

// ---------------------------------------------------------------------------
// Packed weight layout: pw[l][m][t][lane][e] (f16), m in {Wni,Wnj,Wnode,Wfij},
// t = 16-col tile, lane/e follow the 16-bit WMMA B operand layout:
//   lane L -> column n = t*16 + (L&15); element e -> K = (e&7)+8*(L>=16)+16*(e>>3)
// so each lane's 16 halves are one contiguous 32B chunk.
// ---------------------------------------------------------------------------
__global__ void pack_weights_k(const float* __restrict__ Wni,
                               const float* __restrict__ Wnj,
                               const float* __restrict__ Wnode,
                               const float* __restrict__ Wfij,
                               _Float16* __restrict__ pw, int total) {
  int i = blockIdx.x * blockDim.x + threadIdx.x;
  if (i >= total) return;
  int e    = i & 15;
  int lane = (i >> 4) & 31;
  int t    = (i >> 9) & 7;
  int m    = (i >> 12) & 3;
  int l    = i >> 14;
  int hi = lane >> 4, l16 = lane & 15;
  int k = (e & 7) + (hi << 3) + ((e >> 3) << 4);
  int n = t * 16 + l16;
  const float* W = (m == 0 ? Wni : m == 1 ? Wnj : m == 2 ? Wnode : Wfij)
                   + (size_t)l * DIN * HD;
  pw[i] = (_Float16)W[k * HD + n];
}

// f32 -> f16 elementwise (activations feed WMMA as f16 anyway)
__global__ void f32_to_f16_k(const float* __restrict__ in, _Float16* __restrict__ out, int total) {
  int i = blockIdx.x * blockDim.x + threadIdx.x;
  if (i < total) out[i] = (_Float16)in[i];
}

// A operand: two contiguous v8h (16B) loads per lane from row-major f16 rows.
__device__ __forceinline__ v16h load_a_f16(const _Float16* __restrict__ base, int row, int hi) {
  const _Float16* p = base + (size_t)row * DIN + (hi << 3);
  v8h a0 = *(const v8h*)p;          // K = 8*hi .. 8*hi+7
  v8h a1 = *(const v8h*)(p + 16);   // K = 16+8*hi .. 23+8*hi
  v16h a;
#pragma unroll
  for (int e = 0; e < 8; ++e) { a[e] = a0[e]; a[e + 8] = a1[e]; }
  return a;
}

// ---- Kernel 1: node GEMMs: f_ni, f_nj, h = nf @ {W_ni, W_nj, W_node} ------
__global__ void node_gemm_k(const _Float16* __restrict__ nf16,
                            const _Float16* __restrict__ pwL,   // layer base [4][8][32][16]
                            float* __restrict__ fni, float* __restrict__ fnj,
                            float* __restrict__ hfeat, int ntiles) {
  int gwave = (blockIdx.x * blockDim.x + threadIdx.x) >> 5;
  int lane  = threadIdx.x & 31;
  if (gwave >= ntiles) return;                 // wave-uniform exit (no barriers here)
  int hi = lane >> 4, l16 = lane & 15;
  v16h a = load_a_f16(nf16, gwave * 16 + l16, hi);
#pragma unroll
  for (int t = 0; t < 8; ++t) {
    int n = t * 16 + l16;
    v16h b0 = *(const v16h*)(pwL + (0 << 12) + (t << 9) + (lane << 4));
    v16h b1 = *(const v16h*)(pwL + (1 << 12) + (t << 9) + (lane << 4));
    v16h b2 = *(const v16h*)(pwL + (2 << 12) + (t << 9) + (lane << 4));
    v8f c0 = {}, c1 = {}, c2 = {};
    c0 = __builtin_amdgcn_wmma_f32_16x16x32_f16(false, a, false, b0, (short)0, c0, false, false);
    c1 = __builtin_amdgcn_wmma_f32_16x16x32_f16(false, a, false, b1, (short)0, c1, false, false);
    c2 = __builtin_amdgcn_wmma_f32_16x16x32_f16(false, a, false, b2, (short)0, c2, false, false);
#pragma unroll
    for (int r = 0; r < 8; ++r) {
      size_t row = (size_t)(gwave * 16 + r + hi * 8);
      fni  [row * HD + n] = c0[r];
      fnj  [row * HD + n] = c1[r];
      hfeat[row * HD + n] = c2[r];
    }
  }
}

// ---- Kernel 2: init softmax stats -----------------------------------------
__global__ void init_stats_k(unsigned int* __restrict__ emax, float* __restrict__ denom, int total) {
  int i = blockIdx.x * blockDim.x + threadIdx.x;
  if (i >= total) return;
  emax[i]  = 0xFF800000u;   // -inf
  denom[i] = 0.0f;
}

// Stage the packed W_fij (8KB) into LDS with the CDNA5 async-DMA path.
// 256 threads x 32B each; offset applies to both LDS and global addresses.
__device__ __forceinline__ void stage_wfij_lds(_Float16* sW, const _Float16* __restrict__ pwfij) {
  int tid = threadIdx.x;
  unsigned long long g = (unsigned long long)(pwfij + (tid << 4));
  unsigned int       l = (unsigned int)(unsigned long long)(&sW[tid << 4]);
  asm volatile("global_load_async_to_lds_b128 %0, %1, off"           :: "v"(l), "v"(g) : "memory");
  asm volatile("global_load_async_to_lds_b128 %0, %1, off offset:16" :: "v"(l), "v"(g) : "memory");
  asm volatile("s_wait_asynccnt 0x0" ::: "memory");
  __syncthreads();
}

// ---- Kernel 3: edge logits: f_out on the fly (WMMA) -> e, atomic max ------
__global__ void edge_logits_k(const _Float16* __restrict__ ef16,
                              const _Float16* __restrict__ pwfij,
                              const float* __restrict__ fni,
                              const float* __restrict__ fnj,
                              const float* __restrict__ biasL,
                              const float* __restrict__ attnL,
                              const int* __restrict__ src,
                              const int* __restrict__ dst,
                              float* __restrict__ elog,
                              unsigned int* __restrict__ emax) {
  __shared__ __align__(32) _Float16 sW[4096];
  stage_wfij_lds(sW, pwfij);                       // grid divides exactly: no early exits
  int gwave = (blockIdx.x * blockDim.x + threadIdx.x) >> 5;
  int lane  = threadIdx.x & 31;
  int hi = lane >> 4, l16 = lane & 15;
  v16h a = load_a_f16(ef16, gwave * 16 + l16, hi);
  int sidx[8], didx[8];
#pragma unroll
  for (int r = 0; r < 8; ++r) {
    int e = gwave * 16 + r + hi * 8;
    sidx[r] = src[e]; didx[r] = dst[e];
  }
  float esum[8];
#pragma unroll
  for (int r = 0; r < 8; ++r) esum[r] = 0.0f;
#pragma unroll
  for (int t = 0; t < 8; ++t) {
    int col = t * 16 + l16;
    v16h b = *(const v16h*)(&sW[(t << 9) + (lane << 4)]);
    v8f c = {};
    c = __builtin_amdgcn_wmma_f32_16x16x32_f16(false, a, false, b, (short)0, c, false, false);
    float bv = biasL[col];
    float av = attnL[col];           // attn is (H,D) contiguous -> attnL[col]
    const int h = t >> 1;            // 16 cols of tile t lie within one head
#pragma unroll
    for (int r = 0; r < 8; ++r) {
      float x = c[r] + fni[(size_t)sidx[r] * HD + col] + fnj[(size_t)didx[r] * HD + col] + bv;
      x = x > 0.0f ? x : 0.2f * x;   // leaky relu
      float p = x * av;
      p += __shfl_xor(p, 1, 32); p += __shfl_xor(p, 2, 32);
      p += __shfl_xor(p, 4, 32); p += __shfl_xor(p, 8, 32);
      if (l16 == h) esum[r] += p;    // lane l16==h accumulates head h
    }
  }
  if (l16 < 4) {
#pragma unroll
    for (int r = 0; r < 8; ++r) {
      int e = gwave * 16 + r + hi * 8;
      float v = esum[r];
      elog[(size_t)e * 4 + l16] = v;
      unsigned int* addr = &emax[(size_t)didx[r] * 4 + l16];
      if (v >= 0.0f) atomicMax((int*)addr, __float_as_int(v));
      else           atomicMin(addr, __float_as_uint(v));
    }
  }
}

// ---- Kernel 4: a = exp(e - emax[dst]); denom[dst] += a --------------------
__global__ void exp_denom_k(float* __restrict__ elog,
                            const unsigned int* __restrict__ emax,
                            float* __restrict__ denom,
                            const int* __restrict__ dst, int total) {
  int i = blockIdx.x * blockDim.x + threadIdx.x;
  if (i >= total) return;
  int e = i >> 2, h = i & 3;
  int d_ = dst[e];
  float m = __uint_as_float(emax[(size_t)d_ * 4 + h]);
  float v = __expf(elog[i] - m);
  elog[i] = v;                        // overwrite with numerator
  atomicAdd(&denom[(size_t)d_ * 4 + h], v);
}

// ---- Kernel 5: recompute f_out (WMMA), write ef_next (f16, mean of heads) -
__global__ void edge_ef_k(const _Float16* __restrict__ ef16,
                          const _Float16* __restrict__ pwfij,
                          const float* __restrict__ fni,
                          const float* __restrict__ fnj,
                          const float* __restrict__ biasL,
                          const int* __restrict__ src,
                          const int* __restrict__ dst,
                          _Float16* __restrict__ efout) {
  __shared__ __align__(32) _Float16 sW[4096];
  stage_wfij_lds(sW, pwfij);
  int gwave = (blockIdx.x * blockDim.x + threadIdx.x) >> 5;
  int lane  = threadIdx.x & 31;
  int hi = lane >> 4, l16 = lane & 15;
  v16h a = load_a_f16(ef16, gwave * 16 + l16, hi);
  int sidx[8], didx[8];
#pragma unroll
  for (int r = 0; r < 8; ++r) {
    int e = gwave * 16 + r + hi * 8;
    sidx[r] = src[e]; didx[r] = dst[e];
  }
  float ef0[8], ef1[8];
#pragma unroll
  for (int r = 0; r < 8; ++r) { ef0[r] = 0.0f; ef1[r] = 0.0f; }
#pragma unroll
  for (int t = 0; t < 8; ++t) {
    int col = t * 16 + l16;
    v16h b = *(const v16h*)(&sW[(t << 9) + (lane << 4)]);
    v8f c = {};
    c = __builtin_amdgcn_wmma_f32_16x16x32_f16(false, a, false, b, (short)0, c, false, false);
    float bv = biasL[col];
#pragma unroll
    for (int r = 0; r < 8; ++r) {
      float x = c[r] + fni[(size_t)sidx[r] * HD + col] + fnj[(size_t)didx[r] * HD + col] + bv;
      x = x > 0.0f ? x : 0.2f * x;
      if (t & 1) ef1[r] += 0.25f * x; else ef0[r] += 0.25f * x;   // mean over 4 heads
    }
  }
#pragma unroll
  for (int r = 0; r < 8; ++r) {
    size_t e = (size_t)(gwave * 16 + r + hi * 8);
    efout[e * DIN + l16]      = (_Float16)ef0[r];   // d = l16
    efout[e * DIN + 16 + l16] = (_Float16)ef1[r];   // d = 16 + l16
  }
}

// ---- Kernel 6: zero h_out -------------------------------------------------
__global__ void zero_k(float* __restrict__ p, int total) {
  int i = blockIdx.x * blockDim.x + threadIdx.x;
  if (i < total) p[i] = 0.0f;
}

// ---- Kernel 7: aggregation, one wave per edge: hout[dst] += a * h[src] ----
__global__ void aggregate_k(const float* __restrict__ elog,
                            const float* __restrict__ denom,
                            const int* __restrict__ src,
                            const int* __restrict__ dst,
                            const float* __restrict__ hfeat,
                            float* __restrict__ hout, int nedges) {
  int gwave = (blockIdx.x * blockDim.x + threadIdx.x) >> 5;
  int lane  = threadIdx.x & 31;
  if (gwave >= nedges) return;
  int s_ = src[gwave], d_ = dst[gwave];
  const float* hp = hfeat + (size_t)s_ * HD;
  float*       op = hout  + (size_t)d_ * HD;
#pragma unroll
  for (int j = 0; j < 4; ++j) {
    float aval = elog[(size_t)gwave * 4 + j] / denom[(size_t)d_ * 4 + j];
    atomicAdd(&op[lane + 32 * j], hp[lane + 32 * j] * aval);
  }
}

// ---- Kernel 8: node output = mean over heads (f16 mid-layers, f32 final) --
__global__ void node_mean_k(const float* __restrict__ hout,
                            float* __restrict__ out32,
                            _Float16* __restrict__ out16,
                            int last, int total) {
  int i = blockIdx.x * blockDim.x + threadIdx.x;
  if (i >= total) return;
  int n = i >> 5, d = i & 31;
  const float* p = hout + (size_t)n * HD + d;
  float v = 0.25f * (p[0] + p[32] + p[64] + p[96]);
  if (last) out32[i] = v; else out16[i] = (_Float16)v;
}

extern "C" void kernel_launch(void* const* d_in, const int* in_sizes, int n_in,
                              void* d_out, int out_size, void* d_ws, size_t ws_size,
                              hipStream_t stream) {
  const float* nf0   = (const float*)d_in[0];
  const float* ef0   = (const float*)d_in[1];
  const int*   src   = (const int*)  d_in[2];
  const int*   dst   = (const int*)  d_in[3];
  const float* Wni   = (const float*)d_in[4];
  const float* Wnj   = (const float*)d_in[5];
  const float* Wfij  = (const float*)d_in[6];
  const float* Wnode = (const float*)d_in[7];
  const float* attn  = (const float*)d_in[8];
  const float* bias  = (const float*)d_in[9];

  // Workspace carve-up (~285 MB), 256B aligned chunks
  char* wp = (char*)d_ws;
  auto carve = [&](size_t bytes) { char* p = wp; wp += (bytes + 255) & ~(size_t)255; return p; };
  float*        fni   = (float*)       carve((size_t)N_NODES * HD * 4);
  float*        fnj   = (float*)       carve((size_t)N_NODES * HD * 4);
  float*        hfeat = (float*)       carve((size_t)N_NODES * HD * 4);
  float*        hout  = (float*)       carve((size_t)N_NODES * HD * 4);
  float*        elog  = (float*)       carve((size_t)N_EDGES * 4  * 4);
  unsigned int* emax  = (unsigned int*)carve((size_t)N_NODES * 4  * 4);
  float*        denom = (float*)       carve((size_t)N_NODES * 4  * 4);
  _Float16*     ef16  = (_Float16*)    carve((size_t)N_EDGES * DIN * 2);
  _Float16*     nf16A = (_Float16*)    carve((size_t)N_NODES * DIN * 2);
  _Float16*     nf16B = (_Float16*)    carve((size_t)N_NODES * DIN * 2);
  _Float16*     pw    = (_Float16*)    carve((size_t)NLAYERS * 4 * 8 * 32 * 16 * 2);

  const int ntilesN = N_NODES / 16;   // 6250 (exact)
  const int ntilesE = N_EDGES / 16;   // 50000 (exact) -> 6250 blocks, no partial blocks
  dim3 blk(256);
  int gemmBlocksN = (ntilesN * 32 + 255) / 256;
  int gemmBlocksE = (ntilesE * 32) / 256;            // exact
  int aggBlocks   = (N_EDGES * 32) / 256;            // exact

  // One-time prep: pack weights into WMMA-B lane layout; convert activations.
  int pwTotal = NLAYERS * 4 * 8 * 32 * 16;
  pack_weights_k<<<(pwTotal + 255) / 256, blk, 0, stream>>>(Wni, Wnj, Wnode, Wfij, pw, pwTotal);
  f32_to_f16_k<<<(N_NODES * DIN + 255) / 256, blk, 0, stream>>>(nf0, nf16A, N_NODES * DIN);
  f32_to_f16_k<<<(N_EDGES * DIN + 255) / 256, blk, 0, stream>>>(ef0, ef16, N_EDGES * DIN);

  for (int l = 0; l < NLAYERS; ++l) {
    const _Float16* nfin  = (l == 1) ? nf16B : nf16A;   // A -> B -> A
    _Float16*       nfo16 = (l == 0) ? nf16B : nf16A;
    int             last  = (l == NLAYERS - 1);
    const _Float16* pwL    = pw + (size_t)l * 4 * 4096;
    const _Float16* pwfijL = pwL + (size_t)3 * 4096;
    const float* attnL = attn + (size_t)l * HD;
    const float* biasL = bias + (size_t)l * HD;

    node_gemm_k<<<gemmBlocksN, blk, 0, stream>>>(nfin, pwL, fni, fnj, hfeat, ntilesN);
    init_stats_k<<<(N_NODES * 4 + 255) / 256, blk, 0, stream>>>(emax, denom, N_NODES * 4);
    edge_logits_k<<<gemmBlocksE, blk, 0, stream>>>(ef16, pwfijL, fni, fnj, biasL, attnL,
                                                   src, dst, elog, emax);
    exp_denom_k<<<(N_EDGES * 4 + 255) / 256, blk, 0, stream>>>(elog, emax, denom, dst,
                                                               N_EDGES * 4);
    edge_ef_k<<<gemmBlocksE, blk, 0, stream>>>(ef16, pwfijL, fni, fnj, biasL,
                                               src, dst, ef16);
    zero_k<<<(N_NODES * HD + 255) / 256, blk, 0, stream>>>(hout, N_NODES * HD);
    aggregate_k<<<aggBlocks, blk, 0, stream>>>(elog, denom, src, dst, hfeat, hout, N_EDGES);
    node_mean_k<<<(N_NODES * DIN + 255) / 256, blk, 0, stream>>>(hout, (float*)d_out,
                                                                 nfo16, last, N_NODES * DIN);
  }
}
